// RNNPlusGNNModel_45818711113980
// MI455X (gfx1250) — matrix operand order repaired and checked by reading, not compiled, not run on hardware
//
#include <hip/hip_runtime.h>
#include <math.h>

// ---------------- model constants ----------------
#define BB   128
#define TT   512
#define IN_F 256
#define IN_L 320          // IN_F + 64 emb
#define HH   64
#define G4   256          // 4*H
#define MROWS (BB*TT)     // 65536
#define NN   (128*1024)   // nodes
#define EE   (2*1024*1024)
#define NG   128
#define NPG  1024
#define GN   8
#define ATT  64
#define KMAX 320          // max GEMM K (LDS strip sizing)

typedef __attribute__((ext_vector_type(16))) _Float16 v16h;
typedef __attribute__((ext_vector_type(8)))  float    v8f;

// ================= generic helpers =================
__global__ void k_fill(float* p, float v, int n) {
    int i = blockIdx.x * blockDim.x + threadIdx.x;
    if (i < n) p[i] = v;
}

__global__ void k_f32_to_f16(const float* __restrict__ src, _Float16* __restrict__ dst, int n) {
    int i = blockIdx.x * blockDim.x + threadIdx.x;
    if (i < n) dst[i] = (_Float16)src[i];
}

// xcat[m, 0:256] = matrix[m, :], xcat[m, 256:320] = core_emb[core_terms[m], :]
__global__ void k_build_xcat(const float* __restrict__ matrix, const int* __restrict__ ct,
                             const float* __restrict__ emb, _Float16* __restrict__ xcat) {
    long long i = (long long)blockIdx.x * blockDim.x + threadIdx.x;
    if (i >= (long long)MROWS * IN_L) return;
    long long m = i / IN_L;
    int f = (int)(i % IN_L);
    float v = (f < IN_F) ? matrix[m * IN_F + f]
                         : emb[(long long)ct[m] * 64 + (f - IN_F)];
    xcat[i] = (_Float16)v;
}

// ================= WMMA GEMM =================
// C[M,N] = A[M,K] @ W[N,K]^T + bias[N]    (A,W f16 row-major; C f32 row-major)
// Block = 8 waves: 128 output rows x 64 output cols.
//  - 64-row weight strip (contiguous 64*K chunk of W) staged in LDS once per block
//  - each wave owns one 16-row M tile and register-blocks 4 N tiles:
//    per 32-wide K step: 1 A fragment (global) feeds 4 WMMAs (B fragments from LDS)
// Requires: M % 128 == 0, N % 64 == 0, K % 32 == 0, K <= KMAX.
__global__ void __launch_bounds__(256)
k_gemm_wmma(const _Float16* __restrict__ A,
            const _Float16* __restrict__ W,
            const float* __restrict__ bias,
            float* __restrict__ C,
            int M, int N, int K) {
    __shared__ _Float16 Blds[64 * KMAX];      // 40 KB max

    int nstrips = N >> 6;                     // N / 64
    int bm = blockIdx.x / nstrips;
    int bn = blockIdx.x - bm * nstrips;
    int n_base = bn << 6;

    // cooperative flat copy: W rows [n_base, n_base+64) are contiguous
    {
        const _Float16* wsrc = W + (size_t)n_base * K;
        int total = 64 * K;
        for (int i = threadIdx.x; i < total; i += 256) Blds[i] = wsrc[i];
    }
    __syncthreads();

    int lane = threadIdx.x & 31;
    int wave = threadIdx.x >> 5;
    int m0 = (bm << 7) + (wave << 4);
    if (m0 >= M) return;

    int half = lane >> 4;                 // 0: lanes 0-15, 1: lanes 16-31
    int l15  = lane & 15;

    // A 16x32 fragment layout: lane<16 holds K=k0+0..15; lane>=16: k0+16..31
    const _Float16* arow = A + (size_t)(m0 + l15) * K + half * 16;
    // B 32x16 fragment (B[k][n] = W[n][k]): lane<16 holds K=k0+0..15; lane>=16: k0+16..31
    const _Float16* bbase = Blds + (size_t)l15 * K + half * 16;

    v8f acc0 = {}, acc1 = {}, acc2 = {}, acc3 = {};
    for (int k0 = 0; k0 < K; k0 += 32) {
        v16h a;
#pragma unroll
        for (int i = 0; i < 8; ++i) {
            a[i]     = arow[k0 + i];
            a[i + 8] = arow[k0 + 8 + i];
        }
        v16h b0, b1, b2, b3;
#pragma unroll
        for (int i = 0; i < 16; ++i) {
            b0[i] = bbase[k0 + i];
            b1[i] = bbase[16 * K + k0 + i];
            b2[i] = bbase[32 * K + k0 + i];
            b3[i] = bbase[48 * K + k0 + i];
        }
        acc0 = __builtin_amdgcn_wmma_f32_16x16x32_f16(false, a, false, b0, (short)0, acc0, false, false);
        acc1 = __builtin_amdgcn_wmma_f32_16x16x32_f16(false, a, false, b1, (short)0, acc1, false, false);
        acc2 = __builtin_amdgcn_wmma_f32_16x16x32_f16(false, a, false, b2, (short)0, acc2, false, false);
        acc3 = __builtin_amdgcn_wmma_f32_16x16x32_f16(false, a, false, b3, (short)0, acc3, false, false);
    }
    // C/D layout: lane<16 -> N=n0+lane, rows m0+0..7 ; lane>=16 -> rows m0+8..15
    int mb = m0 + half * 8;
    v8f* accp[4] = { &acc0, &acc1, &acc2, &acc3 };
#pragma unroll
    for (int j = 0; j < 4; ++j) {
        int nn = n_base + j * 16 + l15;
        float bv = bias ? bias[nn] : 0.0f;
        v8f av = *accp[j];
#pragma unroll
        for (int r = 0; r < 8; ++r)
            C[(size_t)(mb + r) * N + nn] = av[r] + bv;
    }
}

// ================= LSTM recurrence =================
__device__ __forceinline__ float sigm(float x) { return 1.0f / (1.0f + __expf(-x)); }

// one block per batch row; 256 threads = one gate pre-activation each.
__global__ void k_lstm_seq(const float* __restrict__ gx,   // [B*T, 256] precomputed x-gates(+b)
                           const float* __restrict__ Whh,  // [256, 64] f32
                           float* __restrict__ hf32,       // [B*T, 64] or null
                           _Float16* __restrict__ hf16) {  // [B*T, 64] or null
    int b = blockIdx.x, n = threadIdx.x;
    __shared__ float h[HH], c[HH], gbuf[G4];
    if (n < HH) { h[n] = 0.0f; c[n] = 0.0f; }
    __syncthreads();
    const float* wrow = Whh + n * HH;
    for (int t = 0; t < TT; ++t) {
        float acc = gx[((size_t)b * TT + t) * G4 + n];
#pragma unroll 8
        for (int k = 0; k < HH; ++k) acc += h[k] * wrow[k];
        gbuf[n] = acc;
        __syncthreads();
        if (n < HH) {
            float ig = sigm(gbuf[n]);
            float fg = sigm(gbuf[HH + n]);
            float gg = tanhf(gbuf[2 * HH + n]);
            float og = sigm(gbuf[3 * HH + n]);
            float cn = fg * c[n] + ig * gg;
            c[n] = cn;
            float hn = og * tanhf(cn);
            h[n] = hn;
            size_t idx = ((size_t)b * TT + t) * HH + n;
            if (hf32) hf32[idx] = hn;
            if (hf16) hf16[idx] = (_Float16)hn;
        }
        __syncthreads();
    }
}

// rnn_x[b] = h1[b, length[b]-1] @ Wr^T + br
__global__ void k_last_proj(const float* __restrict__ h1, const int* __restrict__ length,
                            const float* __restrict__ Wr, const float* __restrict__ br,
                            float* __restrict__ rnnx) {
    int b = threadIdx.x;
    if (b >= BB) return;
    const float* hp = h1 + ((size_t)b * TT + (length[b] - 1)) * HH;
    for (int a = 0; a < GN; ++a) {
        float s = br[a];
        for (int j = 0; j < HH; ++j) s += hp[j] * Wr[a * HH + j];
        rnnx[b * GN + a] = s;
    }
}

// ================= GNN =================
__global__ void k_xw_deginit(const float* __restrict__ xin, const float* __restrict__ W,
                             const float* __restrict__ nmask,
                             float* __restrict__ xw, float* __restrict__ deg) {
    int v = blockIdx.x * blockDim.x + threadIdx.x;
    if (v >= NN) return;
    float xi[GN];
#pragma unroll
    for (int f = 0; f < GN; ++f) xi[f] = xin[(size_t)v * GN + f];
#pragma unroll
    for (int o = 0; o < GN; ++o) {
        float s = 0.0f;
#pragma unroll
        for (int f = 0; f < GN; ++f) s += xi[f] * W[o * GN + f];
        xw[(size_t)v * GN + o] = s;
    }
    deg[v] = nmask[v];   // self-loop term of degree
}

__global__ void k_deg_edges(const int* __restrict__ dst, const float* __restrict__ emask,
                            float* __restrict__ deg) {
    int e = blockIdx.x * blockDim.x + threadIdx.x;
    if (e >= EE) return;
    float em = emask[e];
    if (em != 0.0f) atomicAdd(&deg[dst[e]], em);
}

__global__ void k_dinv(const float* __restrict__ deg, float* __restrict__ dinv) {
    int v = blockIdx.x * blockDim.x + threadIdx.x;
    if (v >= NN) return;
    float d = deg[v];
    dinv[v] = (d > 0.0f) ? rsqrtf(fmaxf(d, 1e-12f)) : 0.0f;
}

__global__ void k_scatter(const int* __restrict__ src, const int* __restrict__ dst,
                          const float* __restrict__ emask, const float* __restrict__ dinv,
                          const float* __restrict__ xw, float* __restrict__ agg) {
    int e = blockIdx.x * blockDim.x + threadIdx.x;
    if (e >= EE) return;
    float em = emask[e];
    if (em == 0.0f) return;
    int s = src[e], d = dst[e];
    float cf = em * dinv[s] * dinv[d];
    if (cf == 0.0f) return;
#pragma unroll
    for (int f = 0; f < GN; ++f)
        atomicAdd(&agg[(size_t)d * GN + f], xw[(size_t)s * GN + f] * cf);
}

__global__ void k_node_finish(const float* __restrict__ xw, const float* __restrict__ agg,
                              const float* __restrict__ nmask, const float* __restrict__ dinv,
                              const float* __restrict__ b, float* __restrict__ xg) {
    int v = blockIdx.x * blockDim.x + threadIdx.x;
    if (v >= NN) return;
    float nm = nmask[v], di = dinv[v];
    float sc = nm * di * di;
#pragma unroll
    for (int f = 0; f < GN; ++f) {
        float val = (agg[(size_t)v * GN + f] + xw[(size_t)v * GN + f] * sc + b[f]) * nm;
        xg[(size_t)v * GN + f] = fmaxf(val, 0.0f);
    }
}

// TopKPooling(0.8): one block per graph, one thread per node; count-based stable rank.
__global__ void __launch_bounds__(1024)
k_topk(float* __restrict__ xg, const float* __restrict__ w, float* __restrict__ nmask) {
    int g = blockIdx.x, tid = threadIdx.x;
    int v = g * NPG + tid;
    __shared__ float s_lds[NPG];
    __shared__ float a_lds[NPG];
    float wv[GN], wn = 0.0f;
#pragma unroll
    for (int f = 0; f < GN; ++f) { wv[f] = w[f]; wn += wv[f] * wv[f]; }
    wn = sqrtf(wn);
    float xv[GN], s = 0.0f;
#pragma unroll
    for (int f = 0; f < GN; ++f) { xv[f] = xg[(size_t)v * GN + f]; s += xv[f] * wv[f]; }
    s /= wn;
    float act = nmask[v];
    s_lds[tid] = s; a_lds[tid] = act;
    __syncthreads();
    int rank = 0, nact = 0;
    for (int j = 0; j < NPG; ++j) {
        float sj = s_lds[j];
        if (a_lds[j] > 0.0f) {
            ++nact;
            if (sj > s || (sj == s && j < tid)) ++rank;   // stable-sort tie break
        }
    }
    int k = (4 * nact + 4) / 5;                           // ceil(0.8*nact)
    float keep = (act > 0.0f && rank < k) ? 1.0f : 0.0f;
    float t = tanhf(s) * keep;
#pragma unroll
    for (int f = 0; f < GN; ++f) xg[(size_t)v * GN + f] = xv[f] * t;
    nmask[v] = keep;
}

__global__ void k_emask(const int* __restrict__ src, const int* __restrict__ dst,
                        const float* __restrict__ nmask, float* __restrict__ emask) {
    int e = blockIdx.x * blockDim.x + threadIdx.x;
    if (e >= EE) return;
    float em = emask[e];
    if (em != 0.0f) emask[e] = em * nmask[src[e]] * nmask[dst[e]];
}

__device__ __forceinline__ float blk_red_sum(float v, float* red, int tid) {
    red[tid] = v; __syncthreads();
    for (int st = 128; st > 0; st >>= 1) { if (tid < st) red[tid] += red[tid + st]; __syncthreads(); }
    float r = red[0]; __syncthreads();
    return r;
}

// GlobalAttention pool; out[g] += (sum alpha*x) @ attW^T + (sum alpha)*attb
__global__ void k_attpool(const float* __restrict__ xg, const float* __restrict__ nmask,
                          const float* __restrict__ gW, const float* __restrict__ gb,
                          const float* __restrict__ attW, const float* __restrict__ attb,
                          float* __restrict__ pool) {
    int g = blockIdx.x, tid = threadIdx.x;        // 256 threads, 4 nodes each
    __shared__ float red[256];
    float gw[GN];
#pragma unroll
    for (int f = 0; f < GN; ++f) gw[f] = gW[f];
    float gbv = gb[0];

    float mymax = -1e30f;
    for (int it = 0; it < 4; ++it) {
        int v = g * NPG + it * 256 + tid;
        if (nmask[v] > 0.0f) {
            float gv = gbv;
#pragma unroll
            for (int f = 0; f < GN; ++f) gv += xg[(size_t)v * GN + f] * gw[f];
            mymax = fmaxf(mymax, gv);
        }
    }
    red[tid] = mymax; __syncthreads();
    for (int st = 128; st > 0; st >>= 1) { if (tid < st) red[tid] = fmaxf(red[tid], red[tid + st]); __syncthreads(); }
    float m = red[0]; __syncthreads();

    float se = 0.0f, px[GN];
#pragma unroll
    for (int f = 0; f < GN; ++f) px[f] = 0.0f;
    for (int it = 0; it < 4; ++it) {
        int v = g * NPG + it * 256 + tid;
        if (nmask[v] > 0.0f) {
            float gv = gbv;
#pragma unroll
            for (int f = 0; f < GN; ++f) gv += xg[(size_t)v * GN + f] * gw[f];
            float e = __expf(gv - m);
            se += e;
#pragma unroll
            for (int f = 0; f < GN; ++f) px[f] += e * xg[(size_t)v * GN + f];
        }
    }
    __shared__ float res[1 + GN];
    float st_ = blk_red_sum(se, red, tid);
    if (tid == 0) res[0] = st_;
    for (int f = 0; f < GN; ++f) {
        float pf = blk_red_sum(px[f], red, tid);
        if (tid == 0) res[1 + f] = pf;
    }
    __syncthreads();
    if (tid < ATT) {
        float ssum = res[0];
        float inv = 1.0f / fmaxf(ssum, 1e-12f);
        float asum = ssum * inv;
        float val = asum * attb[tid];
#pragma unroll
        for (int f = 0; f < GN; ++f) val += res[1 + f] * inv * attW[tid * GN + f];
        pool[g * ATT + tid] += val;
    }
}

__global__ void k_final(const float* __restrict__ rnnx, const float* __restrict__ pool,
                        const float* __restrict__ fcW, const float* __restrict__ fcb,
                        float* __restrict__ out) {
    int b = threadIdx.x;
    if (b >= BB) return;
    float s = fcb[0];
    for (int a = 0; a < GN; ++a) s += rnnx[b * GN + a] * fcW[a];
    for (int a = 0; a < ATT; ++a) s += pool[b * ATT + a] * fcW[GN + a];
    out[b] = s;
}

// ================= launcher =================
extern "C" void kernel_launch(void* const* d_in, const int* in_sizes, int n_in,
                              void* d_out, int out_size, void* d_ws, size_t ws_size,
                              hipStream_t stream) {
    const float* matrix   = (const float*)d_in[0];
    const int*   core_t   = (const int*)d_in[1];
    const int*   length   = (const int*)d_in[2];
    const float* x_graph  = (const float*)d_in[3];
    const int*   esrc     = (const int*)d_in[4];
    const int*   edst     = (const int*)d_in[5];
    const float* core_emb = (const float*)d_in[7];
    const float* Wih0     = (const float*)d_in[8];
    const float* Whh0     = (const float*)d_in[9];
    const float* b0       = (const float*)d_in[10];
    const float* Wih1     = (const float*)d_in[11];
    const float* Whh1     = (const float*)d_in[12];
    const float* b1       = (const float*)d_in[13];
    const float* Wr       = (const float*)d_in[14];
    const float* br       = (const float*)d_in[15];
    const float* gcn_W    = (const float*)d_in[16];
    const float* gcn_b    = (const float*)d_in[17];
    const float* topk_w   = (const float*)d_in[18];
    const float* gate_W   = (const float*)d_in[19];
    const float* gate_b   = (const float*)d_in[20];
    const float* att_W    = (const float*)d_in[21];
    const float* att_b    = (const float*)d_in[22];
    const float* fc_W     = (const float*)d_in[23];
    const float* fc_b     = (const float*)d_in[24];
    float* out = (float*)d_out;

    char* ws = (char*)d_ws;
    size_t off = 0;
    auto take = [&](size_t bytes) { char* p = ws + off; off += (bytes + 255) & ~(size_t)255; return p; };
    _Float16* xcat   = (_Float16*)take((size_t)MROWS * IN_L * 2);
    _Float16* wih0h  = (_Float16*)take((size_t)G4 * IN_L * 2);
    _Float16* wih1h  = (_Float16*)take((size_t)G4 * HH * 2);
    float*    gates  = (float*)take((size_t)MROWS * G4 * 4);
    _Float16* h0h    = (_Float16*)take((size_t)MROWS * HH * 2);
    float*    h1f    = (float*)take((size_t)MROWS * HH * 4);
    float*    rnnx   = (float*)take((size_t)BB * GN * 4);
    float*    xg     = (float*)take((size_t)NN * GN * 4);
    float*    xw     = (float*)take((size_t)NN * GN * 4);
    float*    agg    = (float*)take((size_t)NN * GN * 4);
    float*    deg    = (float*)take((size_t)NN * 4);
    float*    dinv   = (float*)take((size_t)NN * 4);
    float*    nmask  = (float*)take((size_t)NN * 4);
    float*    emask  = (float*)take((size_t)EE * 4);
    float*    pool   = (float*)take((size_t)NG * ATT * 4);
    (void)ws_size; (void)in_sizes; (void)n_in; (void)out_size;

    // ---- RNN path ----
    k_f32_to_f16<<<(G4 * IN_L + 255) / 256, 256, 0, stream>>>(Wih0, wih0h, G4 * IN_L);
    k_f32_to_f16<<<(G4 * HH + 255) / 256, 256, 0, stream>>>(Wih1, wih1h, G4 * HH);
    {
        long long tot = (long long)MROWS * IN_L;
        k_build_xcat<<<(int)((tot + 255) / 256), 256, 0, stream>>>(matrix, core_t, core_emb, xcat);
    }
    // layer 0 input GEMM: [65536,320] x [320 -> 256]; grid = (M/128)*(N/64)
    k_gemm_wmma<<<(MROWS / 128) * (G4 / 64), 256, 0, stream>>>(xcat, wih0h, b0, gates, MROWS, G4, IN_L);
    k_lstm_seq<<<BB, G4, 0, stream>>>(gates, Whh0, (float*)nullptr, h0h);
    // layer 1 input GEMM: [65536,64] x [64 -> 256]
    k_gemm_wmma<<<(MROWS / 128) * (G4 / 64), 256, 0, stream>>>(h0h, wih1h, b1, gates, MROWS, G4, HH);
    k_lstm_seq<<<BB, G4, 0, stream>>>(gates, Whh1, h1f, (_Float16*)nullptr);
    k_last_proj<<<1, BB, 0, stream>>>(h1f, length, Wr, br, rnnx);

    // ---- GNN path ----
    k_fill<<<(NN + 255) / 256, 256, 0, stream>>>(nmask, 1.0f, NN);
    k_fill<<<(EE + 255) / 256, 256, 0, stream>>>(emask, 1.0f, EE);
    k_fill<<<(NG * ATT + 255) / 256, 256, 0, stream>>>(pool, 0.0f, NG * ATT);
    for (int i = 0; i < 4; ++i) {
        const float* xin = (i == 0) ? x_graph : xg;
        k_xw_deginit<<<NN / 256, 256, 0, stream>>>(xin, gcn_W + i * GN * GN, nmask, xw, deg);
        k_deg_edges<<<EE / 256, 256, 0, stream>>>(edst, emask, deg);
        k_dinv<<<NN / 256, 256, 0, stream>>>(deg, dinv);
        k_fill<<<(NN * GN + 255) / 256, 256, 0, stream>>>(agg, 0.0f, NN * GN);
        k_scatter<<<EE / 256, 256, 0, stream>>>(esrc, edst, emask, dinv, xw, agg);
        k_node_finish<<<NN / 256, 256, 0, stream>>>(xw, agg, nmask, dinv, gcn_b + i * GN, xg);
        k_topk<<<NG, NPG, 0, stream>>>(xg, topk_w + i * GN, nmask);
        k_emask<<<EE / 256, 256, 0, stream>>>(esrc, edst, nmask, emask);
        k_attpool<<<NG, 256, 0, stream>>>(xg, nmask, gate_W, gate_b, att_W, att_b, pool);
    }

    // ---- combine ----
    k_final<<<1, BB, 0, stream>>>(rnnx, pool, fc_W, fc_b, out);
}
